// NGCF_57071525430156
// MI455X (gfx1250) — matrix operand order repaired and checked by reading, not compiled, not run on hardware
//
#include <hip/hip_runtime.h>

// ---------------------------------------------------------------------------
// NGCF forward for MI455X (gfx1250, wave32, WMMA + TDM).
//
// Math: softmax row-shift invariance makes the conv1d logit path a no-op ->
// coefs = softmax(interaction, axis=1), constant across layers. Dominant cost
// is side0 = norm_adj @ interaction (4096^3 = 137 GFLOP) -> bf16 WMMA
// 16x16x32 with f32 accumulation (interaction is 0/1: exact in bf16).
// bf16 working set (adj, int, int^T, side0 ~ 134 MB) is L2-resident (192 MB),
// so the run is WMMA-throughput bound; HBM sees ~140 MB of conversion traffic
// plus the 71 MB output (~10 us at 23.3 TB/s).
//
// Big GEMM: 128x128 WG tile, 8 waves of 32x64 (8 WMMA / wave / K-step).
// A and B(=int^T) tiles are staged by the Tensor Data Mover (2D D#, bf16,
// pad_interval=16DW/pad_amount=4DW -> padded LDS rows), double-buffered on
// TENSORcnt. Falls back to manual staging if the builtin is absent.
// ---------------------------------------------------------------------------

typedef __attribute__((ext_vector_type(16))) __bf16        v16bf;
typedef __attribute__((ext_vector_type(8)))  float         v8f;
typedef __attribute__((ext_vector_type(4)))  unsigned int  u32x4;
typedef __attribute__((ext_vector_type(2)))  unsigned int  u32x2;
typedef __attribute__((ext_vector_type(4)))  float         f32x4;
typedef __attribute__((ext_vector_type(8)))  int           i32x8;
typedef __attribute__((ext_vector_type(4)))  int           i32x4;
typedef unsigned short u16;

#if defined(__has_builtin)
#if __has_builtin(__builtin_amdgcn_tensor_load_to_lds) && __has_builtin(__builtin_amdgcn_s_wait_tensorcnt)
#define USE_TDM 1
#endif
#endif
#ifndef USE_TDM
#define USE_TDM 0
#endif

static __device__ __forceinline__ float bf2f(u16 h) {
  return __uint_as_float(((unsigned)h) << 16);
}
static __device__ __forceinline__ u16 f2bf(float f) {  // round-to-nearest-even
  unsigned u = __float_as_uint(f);
  unsigned r = u + 0x7FFFu + ((u >> 16) & 1u);
  return (u16)(r >> 16);
}

union Frag  { v16bf v; u32x4 q[2]; };
union Pack8 { u32x4 q; u16 s[8]; };
union Pack4 { u32x2 q; u16 s[4]; };

#if USE_TDM
// D# group 0: count=1 (valid), lds_addr, 57-bit global byte address, type=2.
static __device__ __forceinline__ u32x4 tdm_g0(unsigned lds_off, unsigned long long ga) {
  u32x4 g;
  g[0] = 1u;
  g[1] = lds_off;
  g[2] = (unsigned)(ga & 0xffffffffu);
  g[3] = (unsigned)((ga >> 32) & 0x01ffffffu) | (2u << 30);
  return g;
}
// D# group 1 for a 2D bf16 tile: 32 (dim0=K) x 128 (dim1=rows) out of a
// 4096x4096 tensor, stride0=4096. data_size=1 (2B). LDS padding: after every
// 16 DWORDs (64B row) add 4 DWORDs (16B) -> row stride 80B = 40 u16.
static __device__ __forceinline__ i32x8 tdm_g1_tile() {
  i32x8 g;
  g[0] = (int)((1u << 16) | (1u << 20) | (3u << 22) | (3u << 25));
  g[1] = (int)((4096u & 0xffffu) << 16);        // tensor_dim0[15:0] @63:48
  g[2] = (int)((4096u & 0xffffu) << 16);        // td0[31:16]=0 | td1[15:0] @95:80
  g[3] = (int)(32u << 16);                      // td1[31:16]=0 | tile_dim0 @127:112
  g[4] = (int)128u;                             // tile_dim1=128, tile_dim2=0
  g[5] = (int)4096;                             // stride0[31:0]
  g[6] = 0;                                     // stride0[47:32] | stride1[15:0]
  g[7] = 0;
  return g;
}
static __device__ __forceinline__ void tdm_load(u32x4 g0, i32x8 g1) {
  i32x4 gz4 = {0, 0, 0, 0};
#if __has_include(<hip/amd_detail/amd_gfx1250_TDM.h>)
  i32x8 gz8 = {0, 0, 0, 0, 0, 0, 0, 0};        // 6-arg toolchain (clang-23)
  __builtin_amdgcn_tensor_load_to_lds(g0, g1, gz4, gz4, gz8, 0);
#else
  __builtin_amdgcn_tensor_load_to_lds(g0, g1, gz4, gz4, 0);  // ROCm 7.2, 5-arg
#endif
}
#endif  // USE_TDM

// ---------------------------------------------------------------------------
// Big GEMM: side0 = norm_adj(bf16) @ interaction(bf16), M=N=K=4096.
// A [M,K] row-major; BT [N,K] row-major (interaction pre-transposed) so both
// tiles are contiguous-row 2D TDM loads. Output bf16.
// ---------------------------------------------------------------------------
__global__ __launch_bounds__(256)
void gemm_big(const u16* __restrict__ A, const u16* __restrict__ BT,
              u16* __restrict__ outBF)
{
  __shared__ u16 As[2][128 * 40];
  __shared__ u16 Bs[2][128 * 40];
  const int tid = threadIdx.x;
  const int m0 = blockIdx.y * 128, n0 = blockIdx.x * 128;
  const int w = tid >> 5, lane = tid & 31, lrow = lane & 15, lhi = lane >> 4;
  const int wm = (w & 3) * 32, wn = (w >> 2) * 64;

  v8f acc[2][4] = {};

#if USE_TDM
  const i32x8 g1 = tdm_g1_tile();
  const unsigned aOff[2] = { (unsigned)(size_t)&As[0][0], (unsigned)(size_t)&As[1][0] };
  const unsigned bOff[2] = { (unsigned)(size_t)&Bs[0][0], (unsigned)(size_t)&Bs[1][0] };
  const unsigned long long aG = (unsigned long long)(size_t)(A  + (size_t)m0 * 4096);
  const unsigned long long bG = (unsigned long long)(size_t)(BT + (size_t)n0 * 4096);
  if (w == 0) {                       // TDM: EXEC ignored, one issue per wave
    tdm_load(tdm_g0(aOff[0], aG), g1);
    tdm_load(tdm_g0(bOff[0], bG), g1);
  }
#endif

  for (int it = 0; it < 128; ++it) {
    const int cur = it & 1;
#if USE_TDM
    const int nxt = cur ^ 1;
    if (w == 0) __builtin_amdgcn_s_wait_tensorcnt((short)0);  // buf[cur] landed
    __syncthreads();                   // publish tile; also: buf[nxt] free
    if (w == 0 && it + 1 < 128) {
      unsigned long long off = (unsigned long long)(it + 1) * 64;  // 32 bf16
      tdm_load(tdm_g0(aOff[nxt], aG + off), g1);
      tdm_load(tdm_g0(bOff[nxt], bG + off), g1);
    }
#else
    __syncthreads();
    for (int c = tid; c < 512; c += 256) {   // 128 rows x 4 chunks, A then B
      int r = c >> 2, q = c & 3;
      *(u32x4*)&As[cur][r * 40 + q * 8] =
          *(const u32x4*)(A + (size_t)(m0 + r) * 4096 + it * 32 + q * 8);
      *(u32x4*)&Bs[cur][r * 40 + q * 8] =
          *(const u32x4*)(BT + (size_t)(n0 + r) * 4096 + it * 32 + q * 8);
    }
    __syncthreads();
#endif
    // A frag: lanes<16 M=lrow K0..7,K16..23 ; lanes>=16 K8..15,K24..31
    Frag a0, a1, b[4];
    const int ar0 = (wm + lrow) * 40, ar1 = (wm + 16 + lrow) * 40;
    a0.q[0] = *(const u32x4*)&As[cur][ar0 + lhi * 8];
    a0.q[1] = *(const u32x4*)&As[cur][ar0 + 16 + lhi * 8];
    a1.q[0] = *(const u32x4*)&As[cur][ar1 + lhi * 8];
    a1.q[1] = *(const u32x4*)&As[cur][ar1 + 16 + lhi * 8];
    #pragma unroll
    for (int ni = 0; ni < 4; ++ni) {   // B frag: col N=lrow, K=lhi*8..+7, 16+lhi*8..+7
      const int br = (wn + ni * 16 + lrow) * 40 + lhi * 8;
      b[ni].q[0] = *(const u32x4*)&Bs[cur][br];
      b[ni].q[1] = *(const u32x4*)&Bs[cur][br + 16];
    }
    #pragma unroll
    for (int ni = 0; ni < 4; ++ni) {
      acc[0][ni] = __builtin_amdgcn_wmma_f32_16x16x32_bf16(false, a0.v, false, b[ni].v, (short)0, acc[0][ni], false, false);
      acc[1][ni] = __builtin_amdgcn_wmma_f32_16x16x32_bf16(false, a1.v, false, b[ni].v, (short)0, acc[1][ni], false, false);
    }
  }

  #pragma unroll
  for (int mi = 0; mi < 2; ++mi)
    #pragma unroll
    for (int ni = 0; ni < 4; ++ni)
      #pragma unroll
      for (int v = 0; v < 8; ++v) {    // C layout: M=v+8*(lane>=16), N=lrow
        int row = m0 + wm + mi * 16 + v + lhi * 8;
        int col = n0 + wn + ni * 16 + lrow;
        outBF[(size_t)row * 4096 + col] = f2bf(acc[mi][ni][v]);
      }
}

// ---------------------------------------------------------------------------
// Skinny GEMM: C[M,64] = A[M,K] @ B[K,64], 1 block = 2 waves, 32x64 rows/block.
// MODE: 0 = A, 1 = A.*A2, 2 = exp(A).  EP: 1 f32, 2 +bias f32, 3 attention.
// ---------------------------------------------------------------------------
template<int MODE, int EP>
__global__ __launch_bounds__(64)
void gemm_skinny(const u16* __restrict__ A, const u16* __restrict__ A2,
                 const u16* __restrict__ B, int K,
                 float* __restrict__ outF, u16* __restrict__ outBF,
                 const float* __restrict__ bias,
                 const float* __restrict__ invrs,
                 const float* __restrict__ addmat)
{
  __shared__ u16 As[32 * 40];
  __shared__ u16 Bt[64 * 40];
  const int tid = threadIdx.x;
  const int m0 = blockIdx.y * 32;
  const int w = tid >> 5, lane = tid & 31, lrow = lane & 15, lhi = lane >> 4;
  const int wn = w * 32;

  // hoist staging addresses out of the K loop
  const u16 *aP[2], *a2P[2]; int aL[2];
  #pragma unroll
  for (int i = 0; i < 2; ++i) {
    int c = tid + i * 64, r = c >> 2, q = c & 3;
    aP[i]  = A + (size_t)(m0 + r) * K + q * 8;
    a2P[i] = (MODE == 1) ? (A2 + (size_t)(m0 + r) * K + q * 8) : aP[i];
    aL[i]  = r * 40 + q * 8;
  }
  const u16* bP[4]; int bN8[4], bK[4];
  #pragma unroll
  for (int i = 0; i < 4; ++i) {
    int c = tid + i * 64; bK[i] = c >> 3; bN8[i] = (c & 7) * 8;
    bP[i] = B + (size_t)bK[i] * 64 + bN8[i];
  }

  v8f acc[2][2] = {};
  for (int k0 = 0; k0 < K; k0 += 32) {
    __syncthreads();
    #pragma unroll
    for (int i = 0; i < 2; ++i) {
      u32x4 val = *(const u32x4*)(aP[i] + k0);
      if (MODE == 1) {
        Pack8 a, b, o; a.q = val;
        b.q = *(const u32x4*)(a2P[i] + k0);
        #pragma unroll
        for (int j = 0; j < 8; ++j) o.s[j] = f2bf(bf2f(a.s[j]) * bf2f(b.s[j]));
        val = o.q;
      } else if (MODE == 2) {
        Pack8 a, o; a.q = val;
        #pragma unroll
        for (int j = 0; j < 8; ++j) o.s[j] = f2bf(__expf(bf2f(a.s[j])));
        val = o.q;
      }
      *(u32x4*)&As[aL[i]] = val;
    }
    #pragma unroll
    for (int i = 0; i < 4; ++i) {      // transpose B tile into [n][k]
      Pack8 b; b.q = *(const u32x4*)(bP[i] + (size_t)k0 * 64);
      #pragma unroll
      for (int j = 0; j < 8; ++j) Bt[(bN8[i] + j) * 40 + bK[i]] = b.s[j];
    }
    __syncthreads();
    if (k0 + 32 < K) {
      __builtin_prefetch(aP[0] + k0 + 32, 0, 1);
      __builtin_prefetch(bP[0] + (size_t)(k0 + 32) * 64, 0, 1);
    }
    Frag a0, a1, b0, b1;
    const int ar0 = lrow * 40, ar1 = (16 + lrow) * 40;
    a0.q[0] = *(const u32x4*)&As[ar0 + lhi * 8];
    a0.q[1] = *(const u32x4*)&As[ar0 + 16 + lhi * 8];
    a1.q[0] = *(const u32x4*)&As[ar1 + lhi * 8];
    a1.q[1] = *(const u32x4*)&As[ar1 + 16 + lhi * 8];
    const int br0 = (wn + lrow) * 40 + lhi * 8, br1 = (wn + 16 + lrow) * 40 + lhi * 8;
    b0.q[0] = *(const u32x4*)&Bt[br0];  b0.q[1] = *(const u32x4*)&Bt[br0 + 16];
    b1.q[0] = *(const u32x4*)&Bt[br1];  b1.q[1] = *(const u32x4*)&Bt[br1 + 16];
    acc[0][0] = __builtin_amdgcn_wmma_f32_16x16x32_bf16(false, a0.v, false, b0.v, (short)0, acc[0][0], false, false);
    acc[0][1] = __builtin_amdgcn_wmma_f32_16x16x32_bf16(false, a0.v, false, b1.v, (short)0, acc[0][1], false, false);
    acc[1][0] = __builtin_amdgcn_wmma_f32_16x16x32_bf16(false, a1.v, false, b0.v, (short)0, acc[1][0], false, false);
    acc[1][1] = __builtin_amdgcn_wmma_f32_16x16x32_bf16(false, a1.v, false, b1.v, (short)0, acc[1][1], false, false);
  }

  #pragma unroll
  for (int mi = 0; mi < 2; ++mi)
    #pragma unroll
    for (int ni = 0; ni < 2; ++ni)
      #pragma unroll
      for (int v = 0; v < 8; ++v) {
        int row = m0 + mi * 16 + v + lhi * 8;
        int col = wn + ni * 16 + lrow;
        float x = acc[mi][ni][v];
        if (EP == 1) {
          outF[(size_t)row * 64 + col] = x;
        } else if (EP == 2) {
          outF[(size_t)row * 64 + col] = x + bias[col];
        } else {
          float y = x * invrs[row] + addmat[(size_t)row * 64 + col];
          y = (y > 0.0f) ? y : 0.2f * y;                 // leaky_relu(0.2)
          outF[(size_t)row * 64 + col]  = y;
          outBF[(size_t)row * 64 + col] = f2bf(y);       // next layer's B
        }
      }
}

// ---------------------------------------------------------------------------
__global__ void f32_to_bf16_kernel(const float* __restrict__ s, u16* __restrict__ d, long n) {
  long i = (long)blockIdx.x * blockDim.x + threadIdx.x;
  long stride = (long)gridDim.x * blockDim.x;
  for (; i < n; i += stride) d[i] = f2bf(s[i]);
}

// dst[n][k] = bf16(src[k][n]), 4096x4096, 32x32 tiles through LDS
__global__ __launch_bounds__(256)
void transpose_bf16_kernel(const float* __restrict__ src, u16* __restrict__ dst) {
  __shared__ float tile[32][33];
  const int t = threadIdx.x, r = t >> 3, c4 = (t & 7) * 4;
  const int gr = blockIdx.y * 32, gc = blockIdx.x * 32;
  f32x4 v = *(const f32x4*)(src + (size_t)(gr + r) * 4096 + gc + c4);
  tile[r][c4 + 0] = v[0]; tile[r][c4 + 1] = v[1];
  tile[r][c4 + 2] = v[2]; tile[r][c4 + 3] = v[3];
  __syncthreads();
  Pack4 o;
  #pragma unroll
  for (int j = 0; j < 4; ++j) o.s[j] = f2bf(tile[c4 + j][r]);
  *(u32x2*)(dst + (size_t)(gc + r) * 4096 + gr + c4) = o.q;
}

// invrs[row] = 1 / sum_j exp(interaction[row][j])  (softmax denom; shift-free)
__global__ void rowsum_exp_kernel(const float* __restrict__ x, float* __restrict__ invrs) {
  __shared__ float red[256];
  int row = blockIdx.x, tid = threadIdx.x;
  float s = 0.f;
  for (int j = tid; j < 4096; j += 256) s += __expf(x[(size_t)row * 4096 + j]);
  red[tid] = s; __syncthreads();
  for (int k = 128; k > 0; k >>= 1) { if (tid < k) red[tid] += red[tid + k]; __syncthreads(); }
  if (tid == 0) invrs[row] = 1.0f / red[0];
}

// seq.T from raw reshape: seqT[n*64+c] = bi_flat[c*4096+n], cast to bf16
__global__ void make_seqT_kernel(const float* __restrict__ bi, u16* __restrict__ seqT) {
  int i = blockIdx.x * 256 + threadIdx.x;   // [0, 262144)
  int n = i >> 6, c = i & 63;
  seqT[i] = f2bf(bi[(size_t)c * 4096 + n]);
}

// layers 1-2 (K=64): sum_emb = side@Wg+bg ; bi = (ego.*side)@Wb+bb
__global__ void small_layer_kernel(const float* __restrict__ side, const float* __restrict__ ego,
                                   const float* __restrict__ Wg, const float* __restrict__ bg,
                                   const float* __restrict__ Wb, const float* __restrict__ bb,
                                   float* __restrict__ sum_emb, float* __restrict__ bi) {
  __shared__ float WgL[64 * 64], WbL[64 * 64], sd[256], es[256];
  int tid = threadIdx.x;
  for (int i = tid; i < 4096; i += 256) { WgL[i] = Wg[i]; WbL[i] = Wb[i]; }
  int r = blockIdx.x * 4 + (tid >> 6), c = tid & 63;
  float s = side[(size_t)r * 64 + c];
  sd[tid] = s;
  es[tid] = s * ego[(size_t)r * 64 + c];
  __syncthreads();
  int base = tid & ~63;
  float a1 = 0.f, a2 = 0.f;
  #pragma unroll 8
  for (int k = 0; k < 64; ++k) {
    a1 += sd[base + k] * WgL[k * 64 + c];
    a2 += es[base + k] * WbL[k * 64 + c];
  }
  sum_emb[(size_t)r * 64 + c] = a1 + bg[c];
  bi[(size_t)r * 64 + c]      = a2 + bb[c];
}

// F.normalize(p=2, dim=1) into output slice (out stride 4352)
__global__ void norm_out_kernel(const float* __restrict__ ego, float* __restrict__ out, int colbase) {
  __shared__ float red[256];
  int tid = threadIdx.x;
  int r = blockIdx.x * 4 + (tid >> 6), g = tid & 63;
  float v = ego[(size_t)r * 64 + g];
  red[tid] = v * v; __syncthreads();
  for (int s = 32; s > 0; s >>= 1) { if (g < s) red[tid] += red[tid + s]; __syncthreads(); }
  float nrm = sqrtf(red[tid & ~63]);
  out[(size_t)r * 4352 + colbase + g] = v / fmaxf(nrm, 1e-12f);
}

__global__ void copy_int_kernel(const float* __restrict__ inter, float* __restrict__ out) {
  long i = (long)blockIdx.x * 256 + threadIdx.x;
  long stride = (long)gridDim.x * 256;
  for (; i < 4096L * 4096L; i += stride)
    out[(i >> 12) * 4352 + (i & 4095)] = inter[i];
}
__global__ void copy_fea_kernel(const float* __restrict__ fea, float* __restrict__ out) {
  int i = blockIdx.x * 256 + threadIdx.x;  // 262144
  out[(size_t)(i >> 6) * 4352 + 4288 + (i & 63)] = fea[i];
}

// ---------------------------------------------------------------------------
extern "C" void kernel_launch(void* const* d_in, const int* in_sizes, int n_in,
                              void* d_out, int out_size, void* d_ws, size_t ws_size,
                              hipStream_t stream) {
  (void)in_sizes; (void)n_in; (void)out_size; (void)ws_size;
  const float* norm_adj    = (const float*)d_in[0];
  const float* interaction = (const float*)d_in[1];
  const float* fea         = (const float*)d_in[2];
  const float* W_gc_0 = (const float*)d_in[3];
  const float* b_gc_0 = (const float*)d_in[4];
  const float* W_bi_0 = (const float*)d_in[5];
  const float* b_bi_0 = (const float*)d_in[6];
  const float* W_gc[3] = {nullptr, (const float*)d_in[7],  (const float*)d_in[11]};
  const float* b_gc[3] = {nullptr, (const float*)d_in[8],  (const float*)d_in[12]};
  const float* W_bi[3] = {nullptr, (const float*)d_in[9],  (const float*)d_in[13]};
  const float* b_bi[3] = {nullptr, (const float*)d_in[10], (const float*)d_in[14]};
  // d_in[15..18] (conv path) are provably dead; d_in[19..20] unused by ref.
  float* out = (float*)d_out;

  // workspace carve-out (~140 MB)
  char* w = (char*)d_ws;
  auto alloc = [&](size_t bytes) { char* p = w; w += (bytes + 255) & ~(size_t)255; return p; };
  u16*   adj_bf   = (u16*)alloc(33554432);   // bf16(norm_adj)      [4096,4096]
  u16*   int_bf   = (u16*)alloc(33554432);   // bf16(interaction)   [4096,4096]
  u16*   intT_bf  = (u16*)alloc(33554432);   // bf16(interaction^T) [4096,4096]
  u16*   side0_bf = (u16*)alloc(33554432);   // layer-0 side        [4096,4096]
  u16*   wg0_bf   = (u16*)alloc(524288);
  u16*   wb0_bf   = (u16*)alloc(524288);
  u16*   ego_bf   = (u16*)alloc(524288);
  u16*   seqT     = (u16*)alloc(524288);
  float* invrs    = (float*)alloc(16384);
  float* sum_emb  = (float*)alloc(1048576);
  float* bi       = (float*)alloc(1048576);
  float* ego      = (float*)alloc(1048576);
  float* sidef    = (float*)alloc(1048576);

  // --- precompute ---
  f32_to_bf16_kernel<<<4096, 256, 0, stream>>>(norm_adj,    adj_bf, 16777216L);
  f32_to_bf16_kernel<<<4096, 256, 0, stream>>>(interaction, int_bf, 16777216L);
  transpose_bf16_kernel<<<dim3(128, 128), 256, 0, stream>>>(interaction, intT_bf);
  f32_to_bf16_kernel<<<256,  256, 0, stream>>>(W_gc_0, wg0_bf, 262144L);
  f32_to_bf16_kernel<<<256,  256, 0, stream>>>(W_bi_0, wb0_bf, 262144L);
  rowsum_exp_kernel<<<4096, 256, 0, stream>>>(interaction, invrs);

  // --- layer 0 (ego = interaction) ---
  // side0 = norm_adj @ interaction   (4096^3, TDM-fed WMMA)
  gemm_big<<<dim3(32, 32), 256, 0, stream>>>(adj_bf, intT_bf, side0_bf);
  // sum_emb = side0 @ W_gc_0 + b
  gemm_skinny<0, 2><<<dim3(1, 128), 64, 0, stream>>>(
      side0_bf, nullptr, wg0_bf, 4096, sum_emb, nullptr, b_gc_0, nullptr, nullptr);
  // bi = (interaction .* side0) @ W_bi_0 + b
  gemm_skinny<1, 2><<<dim3(1, 128), 64, 0, stream>>>(
      side0_bf, int_bf, wb0_bf, 4096, bi, nullptr, b_bi_0, nullptr, nullptr);
  make_seqT_kernel<<<1024, 256, 0, stream>>>(bi, seqT);
  // ego = leaky( sum_emb + softmax(interaction) @ seqT , 0.2 )
  gemm_skinny<2, 3><<<dim3(1, 128), 64, 0, stream>>>(
      int_bf, nullptr, seqT, 4096, ego, ego_bf, nullptr, invrs, sum_emb);
  norm_out_kernel<<<1024, 256, 0, stream>>>(ego, out, 4096);

  // --- layers 1, 2 ---
  for (int l = 1; l <= 2; ++l) {
    gemm_skinny<0, 1><<<dim3(1, 128), 64, 0, stream>>>(
        adj_bf, nullptr, ego_bf, 4096, sidef, nullptr, nullptr, nullptr, nullptr);
    small_layer_kernel<<<1024, 256, 0, stream>>>(
        sidef, ego, W_gc[l], b_gc[l], W_bi[l], b_bi[l], sum_emb, bi);
    make_seqT_kernel<<<1024, 256, 0, stream>>>(bi, seqT);
    gemm_skinny<2, 3><<<dim3(1, 128), 64, 0, stream>>>(
        int_bf, nullptr, seqT, 4096, ego, ego_bf, nullptr, invrs, sum_emb);
    norm_out_kernel<<<1024, 256, 0, stream>>>(ego, out, 4096 + 64 * l);
  }

  // --- assemble remaining output columns ---
  copy_int_kernel<<<8192, 256, 0, stream>>>(interaction, out);
  copy_fea_kernel<<<1024, 256, 0, stream>>>(fea, out);
}
